// PointNetSetAbstractionMsg_36172214567330
// MI455X (gfx1250) — matrix operand order repaired and checked by reading, not compiled, not run on hardware
//
#include <hip/hip_runtime.h>
#include <hip/hip_bf16.h>

typedef __attribute__((ext_vector_type(16))) _Float16 v16h;
typedef __attribute__((ext_vector_type(8)))  _Float16 v8h;
typedef __attribute__((ext_vector_type(8)))  float    v8f;

#define BB 4
#define NN 16384
#define SS 1024
#define CIN 64

// ---------------------------------------------------------------------------
// FPS: one block per batch, 1024 threads, 16 points per thread in registers.
// Records centroid index BEFORE the distance update (matches reference scan).
// ---------------------------------------------------------------------------
__global__ void fps_kernel(const float* __restrict__ xyz, int* __restrict__ cent,
                           float* __restrict__ nxyz, float* __restrict__ oxyz) {
    const int b = blockIdx.x, t = threadIdx.x;
    const float* px = xyz + (size_t)b * 3 * NN;
    const float* py = px + NN;
    const float* pz = py + NN;
    float X[16], Y[16], Z[16], D[16];
#pragma unroll
    for (int i = 0; i < 16; ++i) {
        int n = i * 1024 + t;
        X[i] = px[n]; Y[i] = py[n]; Z[i] = pz[n]; D[i] = 1e10f;
    }
    __shared__ float cxyz[3];
    __shared__ float wd[32];
    __shared__ int   wi[32];
    __shared__ int   s_far;
    int far = 0;
    for (int s = 0; s < SS; ++s) {
        if (t == 0) {
            float cx = px[far], cy = py[far], cz = pz[far];
            cxyz[0] = cx; cxyz[1] = cy; cxyz[2] = cz;
            cent[b * SS + s] = far;
            int bs = b * SS + s;
            nxyz[bs * 3 + 0] = cx; nxyz[bs * 3 + 1] = cy; nxyz[bs * 3 + 2] = cz;
            oxyz[(size_t)b * 3 * SS + 0 * SS + s] = cx;
            oxyz[(size_t)b * 3 * SS + 1 * SS + s] = cy;
            oxyz[(size_t)b * 3 * SS + 2 * SS + s] = cz;
        }
        __syncthreads();
        float lcx = cxyz[0], lcy = cxyz[1], lcz = cxyz[2];
        float bd = -1.0f; int bi = 0;
#pragma unroll
        for (int i = 0; i < 16; ++i) {
            float dx = X[i] - lcx, dy = Y[i] - lcy, dz = Z[i] - lcz;
            float d = dx * dx + dy * dy + dz * dz;
            D[i] = fminf(D[i], d);
            if (D[i] > bd) { bd = D[i]; bi = i * 1024 + t; }
        }
        // wave32 argmax reduce
        for (int off = 16; off > 0; off >>= 1) {
            float od = __shfl_down(bd, off);
            int   oi = __shfl_down(bi, off);
            if (od > bd) { bd = od; bi = oi; }
        }
        if ((t & 31) == 0) { wd[t >> 5] = bd; wi[t >> 5] = bi; }
        __syncthreads();
        if (t == 0) {
            float m = wd[0]; int mi = wi[0];
            for (int w = 1; w < 32; ++w) if (wd[w] > m) { m = wd[w]; mi = wi[w]; }
            s_far = mi;
        }
        __syncthreads();
        far = s_far;
    }
}

// ---------------------------------------------------------------------------
// Ball query: one WAVE per centroid. Ballot + prefix-popcount compaction keeps
// indices in ascending order (lanes ordered in chunk, chunks sequential).
// ---------------------------------------------------------------------------
__global__ void ball_query(const float* __restrict__ xyz, const float* __restrict__ nxyz,
                           int* __restrict__ gidx, float r2, int K) {
    const int wid  = (blockIdx.x * blockDim.x + threadIdx.x) >> 5;  // centroid
    const int lane = threadIdx.x & 31;
    if (wid >= BB * SS) return;
    const int b = wid / SS;
    const float* px = xyz + (size_t)b * 3 * NN;
    const float* py = px + NN;
    const float* pz = py + NN;
    const float cx = nxyz[wid * 3 + 0], cy = nxyz[wid * 3 + 1], cz = nxyz[wid * 3 + 2];
    int* out = gidx + (size_t)wid * K;
    int base = 0;
    int first = -1;
    for (int n0 = 0; n0 < NN && base < K; n0 += 32) {
        const int n = n0 + lane;
        float dx = px[n] - cx, dy = py[n] - cy, dz = pz[n] - cz;
        bool pred = (dx * dx + dy * dy + dz * dz) <= r2;
        unsigned mask = (unsigned)__ballot((int)pred);
        if (first < 0 && mask) first = n0 + (__ffs(mask) - 1);
        if (pred) {
            int pos = base + __popc(mask & ((1u << lane) - 1u));
            if (pos < K) out[pos] = n;
        }
        base += __popc(mask);
    }
    if (first < 0) first = 0;
    int cnt = base < K ? base : K;
    for (int i = cnt + lane; i < K; i += 32) out[i] = first;
}

// ---------------------------------------------------------------------------
// Group + pack to f16, column-major [col][Kpad], element-parallel (coalesced
// writes): ch 0..63 = points, 64..66 = relative xyz, rest zero pad.
// col = (b*S + s)*K + k.
// ---------------------------------------------------------------------------
__global__ void group_pack(const float* __restrict__ xyz, const float* __restrict__ pts,
                           const float* __restrict__ nxyz, const int* __restrict__ gidx,
                           _Float16* __restrict__ X, int K, int Kpad, int ncols) {
    size_t idx = (size_t)blockIdx.x * blockDim.x + threadIdx.x;
    size_t total = (size_t)ncols * Kpad;
    if (idx >= total) return;
    int c = (int)(idx % Kpad);
    int col = (int)(idx / Kpad);
    int bs = col / K;
    int b = bs / SS;
    float v = 0.f;
    if (c < CIN) {
        int n = gidx[col];
        v = pts[(size_t)b * CIN * NN + (size_t)c * NN + n];
    } else if (c < CIN + 3) {
        int n = gidx[col];
        int d = c - CIN;
        v = xyz[(size_t)b * 3 * NN + (size_t)d * NN + n] - nxyz[(size_t)bs * 3 + d];
    }
    X[idx] = (_Float16)v;
}

// ---------------------------------------------------------------------------
// Weight pack: f32 [Cout][Cin] -> f16 [Cout][Kpad] zero-padded.
// ---------------------------------------------------------------------------
__global__ void pack_w(const float* __restrict__ w, _Float16* __restrict__ wp,
                       int Cout, int Cin, int Kpad) {
    int i = blockIdx.x * blockDim.x + threadIdx.x;
    if (i >= Cout * Kpad) return;
    int m = i / Kpad, c = i % Kpad;
    wp[i] = (c < Cin) ? (_Float16)w[m * Cin + c] : (_Float16)0.f;
}

// ---------------------------------------------------------------------------
// WMMA GEMM: Y[col][MT*16] = W[MT*16][KB*32] * X[col][KB*32]
// One wave per 16-column tile; all M tiles per wave; f32 accumulate.
// B frag: lane<16 -> K 0..15 of col j0+lane; lane>=16 -> K 16..31 (contiguous).
// A frag: lane group selects K half (0/16), two contiguous 8-f16 chunks.
// D: VGPR r -> M = r + (lane>=16?8:0), N = lane&15.
// ---------------------------------------------------------------------------
template <int MT, int KB>
__global__ void gemm_wmma(const _Float16* __restrict__ X, const _Float16* __restrict__ W,
                          float* __restrict__ Y, int ncols) {
    const int lane = threadIdx.x & 31;
    const int wave = threadIdx.x >> 5;
    const int j0 = (blockIdx.x * (blockDim.x >> 5) + wave) * 16;
    if (j0 >= ncols) return;
    const int Kpad = KB * 32;
    const int colAB = j0 + (lane & 15);
    const int kHalf = (lane >> 4) ? 16 : 0;  // B fragment K base
    const int aHalf = (lane >> 4) ? 16 : 0;  // A fragment K base
    const int am = lane & 15;

    v8f acc[MT];
#pragma unroll
    for (int mt = 0; mt < MT; ++mt)
#pragma unroll
        for (int i = 0; i < 8; ++i) acc[mt][i] = 0.f;

#pragma unroll
    for (int kb = 0; kb < KB; ++kb) {
        // warm cache for the next column tile of the streaming operand
        __builtin_prefetch(X + (size_t)(colAB + 16) * Kpad + kb * 32 + kHalf, 0, 3);
        v16h bfrag = *(const v16h*)(X + (size_t)colAB * Kpad + kb * 32 + kHalf);
#pragma unroll
        for (int mt = 0; mt < MT; ++mt) {
            const _Float16* wrow = W + (size_t)(mt * 16 + am) * Kpad + kb * 32;
            v8h alo = *(const v8h*)(wrow + aHalf);
            v8h ahi = *(const v8h*)(wrow + 8 + aHalf);
            v16h afrag = __builtin_shufflevector(alo, ahi, 0, 1, 2, 3, 4, 5, 6, 7,
                                                 8, 9, 10, 11, 12, 13, 14, 15);
            acc[mt] = __builtin_amdgcn_wmma_f32_16x16x32_f16(
                false, afrag, false, bfrag, (short)0, acc[mt], false, false);
        }
    }
    const int mBase = (lane >> 4) ? 8 : 0;
#pragma unroll
    for (int mt = 0; mt < MT; ++mt) {
        float* yp = Y + (size_t)colAB * (MT * 16) + mt * 16 + mBase;
        *(v8f*)yp = acc[mt];
    }
}

// ---------------------------------------------------------------------------
// BN statistics, coalesced: grid-stride over elements, LDS-atomic per-channel
// accumulation (consecutive lanes -> consecutive channels, no address clash
// within a wave since Cout >= 32), one global atomic per channel per block.
// ---------------------------------------------------------------------------
__global__ void zero_f32(float* __restrict__ p, int n) {
    int i = blockIdx.x * blockDim.x + threadIdx.x;
    if (i < n) p[i] = 0.f;
}

__global__ void bn_stats_acc(const float* __restrict__ Y, int ncols, int Cout,
                             float* __restrict__ acc) {
    __shared__ float ls[256];  // [sum(Cout) | sumsq(Cout)], Cout <= 128
    const int t = threadIdx.x;
    for (int i = t; i < 2 * Cout; i += blockDim.x) ls[i] = 0.f;
    __syncthreads();
    size_t total = (size_t)ncols * Cout;
    size_t stride = (size_t)gridDim.x * blockDim.x;
    for (size_t i = (size_t)blockIdx.x * blockDim.x + t; i < total; i += stride) {
        int m = (int)(i % (size_t)Cout);
        float v = Y[i];
        atomicAdd(&ls[m], v);
        atomicAdd(&ls[Cout + m], v * v);
    }
    __syncthreads();
    for (int i = t; i < 2 * Cout; i += blockDim.x) atomicAdd(&acc[i], ls[i]);
}

__global__ void bn_stats_final(const float* __restrict__ acc, float* __restrict__ st,
                               int ncols, int Cout) {
    int m = threadIdx.x;
    if (m >= Cout) return;
    float mean = acc[m] / (float)ncols;
    float var = acc[Cout + m] / (float)ncols - mean * mean;
    st[m] = mean;
    st[Cout + m] = rsqrtf(var + 1e-5f);
}

// ---------------------------------------------------------------------------
// BN + ReLU + repack to f16 [col][KpadN], element-parallel (coalesced).
// ---------------------------------------------------------------------------
__global__ void bn_relu_pack(const float* __restrict__ Y, const float* __restrict__ st,
                             const float* __restrict__ gamma, const float* __restrict__ beta,
                             _Float16* __restrict__ Xn, int ncols, int Cout, int KpadN) {
    size_t idx = (size_t)blockIdx.x * blockDim.x + threadIdx.x;
    size_t total = (size_t)ncols * KpadN;
    if (idx >= total) return;
    int m = (int)(idx % KpadN);
    int col = (int)(idx / KpadN);
    float v = 0.f;
    if (m < Cout) {
        v = (Y[(size_t)col * Cout + m] - st[m]) * st[Cout + m] * gamma[m] + beta[m];
        v = fmaxf(v, 0.f);
    }
    Xn[idx] = (_Float16)v;
}

// ---------------------------------------------------------------------------
// Final layer: BN + ReLU + max over K, write to output [b][chBase+m][s].
// ---------------------------------------------------------------------------
__global__ void bn_relu_max(const float* __restrict__ Y, const float* __restrict__ st,
                            const float* __restrict__ gamma, const float* __restrict__ beta,
                            float* __restrict__ out, int K, int Cout, int chBase) {
    int idx = blockIdx.x * blockDim.x + threadIdx.x;
    if (idx >= BB * SS * Cout) return;
    int m = idx % Cout;
    int bs = idx / Cout;
    int b = bs / SS, s = bs % SS;
    float mu = st[m], rs = st[Cout + m], g = gamma[m], be = beta[m];
    float mx = 0.f;
    for (int k = 0; k < K; ++k) {
        float v = (Y[((size_t)bs * K + k) * Cout + m] - mu) * rs * g + be;
        mx = fmaxf(mx, fmaxf(v, 0.f));
    }
    out[(size_t)b * 320 * SS + (size_t)(chBase + m) * SS + s] = mx;
}

// ---------------------------------------------------------------------------
static inline void launch_gemm(int MT, int KB, const _Float16* X, const _Float16* W,
                               float* Y, int ncols, hipStream_t st) {
    dim3 g(ncols / 128), bl(256);
#define GEMM_CASE(mt, kb)                                         \
    if (MT == mt && KB == kb) {                                   \
        gemm_wmma<mt, kb><<<g, bl, 0, st>>>(X, W, Y, ncols);      \
        return;                                                   \
    }
    GEMM_CASE(2, 1) GEMM_CASE(2, 2) GEMM_CASE(2, 3)
    GEMM_CASE(4, 1) GEMM_CASE(4, 2) GEMM_CASE(4, 3)
    GEMM_CASE(6, 2) GEMM_CASE(6, 3)
    GEMM_CASE(8, 2) GEMM_CASE(8, 3)
#undef GEMM_CASE
}

extern "C" void kernel_launch(void* const* d_in, const int* in_sizes, int n_in,
                              void* d_out, int out_size, void* d_ws, size_t ws_size,
                              hipStream_t stream) {
    (void)in_sizes; (void)n_in; (void)out_size; (void)ws_size;
    const float* xyz = (const float*)d_in[0];
    const float* pts = (const float*)d_in[1];
    auto Wp_in = [&](int sc, int l) { return (const float*)d_in[2 + (sc * 3 + l) * 4 + 0]; };
    auto G_in  = [&](int sc, int l) { return (const float*)d_in[2 + (sc * 3 + l) * 4 + 2]; };
    auto Be_in = [&](int sc, int l) { return (const float*)d_in[2 + (sc * 3 + l) * 4 + 3]; };

    char* ws = (char*)d_ws;
    auto al = [](size_t x) { return (x + 255) & ~(size_t)255; };
    size_t o = 0;
    int*   cent  = (int*)(ws + o);   o = al(o + (size_t)BB * SS * 4);
    float* nxyz  = (float*)(ws + o); o = al(o + (size_t)BB * SS * 3 * 4);
    float* stacc = (float*)(ws + o); o = al(o + 2 * 128 * 4);
    float* stats = (float*)(ws + o); o = al(o + 2 * 128 * 4);
    _Float16* Wpk = (_Float16*)(ws + o); o = al(o + (size_t)128 * 96 * 2);
    int*   gidx  = (int*)(ws + o);   o = al(o + (size_t)BB * SS * 128 * 4);
    const size_t maxCols = (size_t)BB * SS * 128;
    _Float16* Xa = (_Float16*)(ws + o); o = al(o + maxCols * 96 * 2);
    _Float16* Xb = (_Float16*)(ws + o); o = al(o + maxCols * 96 * 2);
    float* Yb = (float*)(ws + o); o = al(o + maxCols * 128 * 4);

    float* out = (float*)d_out;
    float* out_pts = out + (size_t)BB * 3 * SS;

    const int   Kn[3]      = {16, 32, 128};
    const float r2v[3]     = {0.1f * 0.1f, 0.2f * 0.2f, 0.4f * 0.4f};
    const int   mlps[3][3] = {{32, 32, 64}, {64, 64, 128}, {64, 96, 128}};
    const int   chBase[3]  = {0, 64, 192};

    fps_kernel<<<BB, 1024, 0, stream>>>(xyz, cent, nxyz, out);

    for (int sc = 0; sc < 3; ++sc) {
        const int K = Kn[sc];
        const int ncols = BB * SS * K;
        ball_query<<<(BB * SS * 32 + 255) / 256, 256, 0, stream>>>(xyz, nxyz, gidx, r2v[sc], K);
        {
            size_t total = (size_t)ncols * 96;
            group_pack<<<(unsigned)((total + 255) / 256), 256, 0, stream>>>(
                xyz, pts, nxyz, gidx, Xa, K, 96, ncols);
        }

        _Float16* Xc = Xa;
        _Float16* Xn = Xb;
        int Cin = 67;
        for (int l = 0; l < 3; ++l) {
            const int Cout = mlps[sc][l];
            const int Kpad = ((Cin + 31) / 32) * 32;
            pack_w<<<(Cout * Kpad + 255) / 256, 256, 0, stream>>>(Wp_in(sc, l), Wpk, Cout, Cin, Kpad);
            launch_gemm(Cout / 16, Kpad / 32, Xc, Wpk, Yb, ncols, stream);
            zero_f32<<<1, 256, 0, stream>>>(stacc, 2 * Cout);
            bn_stats_acc<<<1024, 256, 0, stream>>>(Yb, ncols, Cout, stacc);
            bn_stats_final<<<1, 256, 0, stream>>>(stacc, stats, ncols, Cout);
            if (l < 2) {
                const int KpadN = ((Cout + 31) / 32) * 32;
                size_t total = (size_t)ncols * KpadN;
                bn_relu_pack<<<(unsigned)((total + 255) / 256), 256, 0, stream>>>(
                    Yb, stats, G_in(sc, l), Be_in(sc, l), Xn, ncols, Cout, KpadN);
                _Float16* t = Xc; Xc = Xn; Xn = t;
            } else {
                bn_relu_max<<<(BB * SS * Cout + 255) / 256, 256, 0, stream>>>(
                    Yb, stats, G_in(sc, l), Be_in(sc, l), out_pts, K, Cout, chBase[sc]);
            }
            Cin = Cout;
        }
    }
}
